// Llama3_47055661695219
// MI455X (gfx1250) — compile-verified, matrix operand order checked by reading, not run on hardware
//
#include <hip/hip_runtime.h>
#include <hip/hip_bf16.h>
#include <math.h>

#define DIM 4096
#define NHEADS 32
#define NKV 8
#define HDIM 128
#define FFN 14336
#define VOCAB 32000
#define SEQ 2048
#define RMS_EPS 1e-5f
#define ROPE_THETA 500000.0f

typedef unsigned short u16;
typedef __attribute__((ext_vector_type(16))) __bf16 v16bf;
typedef __attribute__((ext_vector_type(16))) unsigned short v16u;
typedef __attribute__((ext_vector_type(8)))  float v8f;

// ---- async global->LDS copy path (ASYNCcnt), gated on builtin presence ----
#if defined(__has_builtin)
#if __has_builtin(__builtin_amdgcn_global_load_async_to_lds_b128)
#define HAVE_ASYNC 1
#endif
#endif
#ifndef HAVE_ASYNC
#define HAVE_ASYNC 0
#endif

// Builtin signature (from hipcc diagnostic): param0 = int4* in global (AS1)
// address space, param1 = int4* in LDS (AS3), then imm offset + imm cpol.
typedef int v4i __attribute__((vector_size(16)));
typedef __attribute__((address_space(1))) v4i* g128p;
typedef __attribute__((address_space(3))) v4i* l128p;

__device__ __forceinline__ void copy16B(const u16* g, u16* l) {
#if HAVE_ASYNC
  u16* gg = const_cast<u16*>(g);
  __builtin_amdgcn_global_load_async_to_lds_b128((g128p)gg, (l128p)l, 0, 0);
#else
  *(uint4*)l = *(const uint4*)g;
#endif
}

__device__ __forceinline__ void async_wait_all() {
#if HAVE_ASYNC
#if __has_builtin(__builtin_amdgcn_s_wait_asynccnt)
  __builtin_amdgcn_s_wait_asynccnt(0);
#else
  asm volatile("s_wait_asynccnt 0" ::: "memory");
#endif
#endif
}

__device__ __forceinline__ u16 f2bf(float f) {
  unsigned int u = __builtin_bit_cast(unsigned int, f);
  u += 0x7fffu + ((u >> 16) & 1u);           // round-to-nearest-even
  return (u16)(u >> 16);
}

// Load a 16-bf16 WMMA fragment: two contiguous 16B chunks at p and p+16
// (ISA A/B layout: per-lane K = {kb..kb+7, kb+16..kb+23}).
__device__ __forceinline__ v16bf ldfrag(const u16* p) {
  uint4 lo = *(const uint4*)(p);
  uint4 hi = *(const uint4*)(p + 16);
  v16u t;
  ((uint4*)&t)[0] = lo;
  ((uint4*)&t)[1] = hi;
  return __builtin_bit_cast(v16bf, t);
}

__device__ __forceinline__ v8f wmma_bf16(v16bf a, v16bf b, v8f c) {
  return __builtin_amdgcn_wmma_f32_16x16x32_bf16(false, a, false, b, (short)0, c,
                                                 false, false);
}

// ---------------------------------------------------------------------------
// bf16 GEMM: C[M,N] (f32) = A[M,K] (bf16) * W (f32 -> bf16 during staging).
// NK=true: W stored [N,K] (apply W^T).  NK=false: W stored [K,N].
// BM=BN=128, BK=64, double-buffered LDS. 256 thr = 8 waves (2Mx4N); each wave
// owns 64x32 = 4x2 WMMA tiles, 16 WMMAs per k-stage between barriers.
// ---------------------------------------------------------------------------
#define LDG 72   // GEMM LDS row stride in ushorts (144B, 16B aligned, skewed)

template <bool NK>
__global__ __launch_bounds__(256) void k_gemm(const u16* __restrict__ A,
                                              const float* __restrict__ W,
                                              float* __restrict__ C,
                                              int M, int N, int K) {
  __shared__ u16 As[2][128 * LDG];   // As[buf][row][k]
  __shared__ u16 Bs[2][128 * LDG];   // Bs[buf][n][k]

  const int tid  = threadIdx.x;
  const int wid  = tid >> 5;
  const int lane = tid & 31;
  const int half = lane >> 4;
  const int lh   = lane & 15;
  const int kb   = half * 8;
  const int wm   = wid >> 2;       // 0..1
  const int wn   = wid & 3;        // 0..3
  const int m0   = blockIdx.y * 128;
  const int n0   = blockIdx.x * 128;

  const v8f vz = {0.f, 0.f, 0.f, 0.f, 0.f, 0.f, 0.f, 0.f};
  v8f acc[4][2];
#pragma unroll
  for (int mt = 0; mt < 4; mt++)
#pragma unroll
    for (int nt = 0; nt < 2; nt++) acc[mt][nt] = vz;

  // staging roles
  const int rowA = tid >> 1, segA = (tid & 1) * 32;   // A: 128 rows x 64 k
  const int nB   = tid >> 1, segB = (tid & 1) * 32;   // W[N,K]: 128 n x 64 k
  const int k0c  = tid >> 3,       g0c = (tid & 7) * 16;        // W[K,N] chunk 0
  const int k1c  = (tid + 256) >> 3, g1c = ((tid + 256) & 7) * 16;  // chunk 1

  auto stageA = [&](int kk, int buf) {
    const u16* g = A + (size_t)(m0 + rowA) * K + kk + segA;
    u16* l = &As[buf][rowA * LDG + segA];
#pragma unroll
    for (int j = 0; j < 4; j++) copy16B(g + j * 8, l + j * 8);
  };
  auto loadB = [&](int kk, float4* wr) {
    if (NK) {
      const float* gw = W + (size_t)(n0 + nB) * K + kk + segB;
#pragma unroll
      for (int j = 0; j < 8; j++) wr[j] = ((const float4*)gw)[j];
      if (kk + 64 < K) __builtin_prefetch(gw + 64, 0, 3);
    } else {
      const float* g0 = W + (size_t)(kk + k0c) * N + n0 + g0c;
      const float* g1 = W + (size_t)(kk + k1c) * N + n0 + g1c;
#pragma unroll
      for (int j = 0; j < 4; j++) wr[j] = ((const float4*)g0)[j];
#pragma unroll
      for (int j = 0; j < 4; j++) wr[4 + j] = ((const float4*)g1)[j];
      if (kk + 64 < K) __builtin_prefetch(g0 + (size_t)64 * N, 0, 3);
    }
  };
  auto storeB = [&](const float4* wr, int buf) {
    if (NK) {
      const float* wf = (const float*)wr;
      uint4 p0, p1, p2, p3;
      unsigned* pu = (unsigned*)&p0;
#pragma unroll
      for (int j = 0; j < 16; j++)
        pu[j] = (unsigned)f2bf(wf[2 * j]) | ((unsigned)f2bf(wf[2 * j + 1]) << 16);
      uint4* l = (uint4*)&Bs[buf][nB * LDG + segB];
      l[0] = p0; l[1] = p1; l[2] = p2; l[3] = p3;
    } else {
      const float* wf = (const float*)wr;
#pragma unroll
      for (int i = 0; i < 16; i++) Bs[buf][(g0c + i) * LDG + k0c] = f2bf(wf[i]);
#pragma unroll
      for (int i = 0; i < 16; i++) Bs[buf][(g1c + i) * LDG + k1c] = f2bf(wf[16 + i]);
    }
  };

  // ---- prologue: fill buffer 0 ----
  {
    stageA(0, 0);
    float4 wr[8];
    loadB(0, wr);
    storeB(wr, 0);
    async_wait_all();
  }
  __syncthreads();

  const int iters = K / 64;
  for (int it = 0; it < iters; ++it) {
    const int cur = it & 1, nxt = cur ^ 1;
    const bool more = (it + 1 < iters);
    float4 wr[8];
    if (more) {
      stageA((it + 1) * 64, nxt);     // async copies fly under compute
      loadB((it + 1) * 64, wr);       // W loads in flight during WMMAs
    }
    // ---- compute current buffer: 2 k-steps x 8 WMMAs ----
#pragma unroll
    for (int s = 0; s < 2; s++) {
      v16bf af[4], bfr[2];
#pragma unroll
      for (int mt = 0; mt < 4; mt++)
        af[mt] = ldfrag(&As[cur][(wm * 64 + mt * 16 + lh) * LDG + s * 32 + kb]);
#pragma unroll
      for (int nt = 0; nt < 2; nt++)
        bfr[nt] = ldfrag(&Bs[cur][(wn * 32 + nt * 16 + lh) * LDG + s * 32 + kb]);
#pragma unroll
      for (int mt = 0; mt < 4; mt++)
#pragma unroll
        for (int nt = 0; nt < 2; nt++)
          acc[mt][nt] = wmma_bf16(af[mt], bfr[nt], acc[mt][nt]);
    }
    if (more) {
      storeB(wr, nxt);
      async_wait_all();
    }
    __syncthreads();
  }

  // ---- store C (f32): element (r, lane) -> row r+8*half, col lh ----
#pragma unroll
  for (int mt = 0; mt < 4; mt++)
#pragma unroll
    for (int nt = 0; nt < 2; nt++) {
      const int col = n0 + wn * 32 + nt * 16 + lh;
#pragma unroll
      for (int r = 0; r < 8; r++) {
        const int row = m0 + wm * 64 + mt * 16 + r + 8 * half;
        C[(size_t)row * N + col] = acc[mt][nt][r];
      }
    }
}

// ---------------------------------------------------------------------------
// Flash attention (bf16 WMMA, fp32 online softmax).
// grid = (SEQ/128, NHEADS). 256 thr = 8 waves; wave w owns q rows
// [blk*128 + w*16, +16). KV streamed in 32-key chunks staged in LDS.
// ---------------------------------------------------------------------------
#define LDST 48

__global__ __launch_bounds__(256) void k_flash(const u16* __restrict__ Q,
                                               const u16* __restrict__ Kc,
                                               const u16* __restrict__ Vc,
                                               u16* __restrict__ O) {
  __shared__ u16 Ks[32 * 136];        // Ks[key][d]
  __shared__ u16 Vt[128 * LDST];      // Vt[d][key]
  __shared__ u16 Ps[8][16 * LDST];    // per-wave P tile [row][key]

  const int tid  = threadIdx.x;
  const int wid  = tid >> 5;
  const int lane = tid & 31;
  const int half = lane >> 4;
  const int lh   = lane & 15;
  const int kb   = half * 8;
  const int h    = blockIdx.y;
  const int kvh  = h >> 2;            // GROUP = 4
  const int qbase = blockIdx.x * 128 + wid * 16;
  const float scale = 0.08838834764831845f;  // 1/sqrt(128)

  v16bf qa[4];
#pragma unroll
  for (int ks = 0; ks < 4; ks++)
    qa[ks] = ldfrag(Q + (size_t)(qbase + lh) * DIM + h * HDIM + ks * 32 + kb);

  const v8f vz = {0.f, 0.f, 0.f, 0.f, 0.f, 0.f, 0.f, 0.f};
  v8f o[8];
#pragma unroll
  for (int nt = 0; nt < 8; nt++) o[nt] = vz;
  float m[8], l[8];
#pragma unroll
  for (int r = 0; r < 8; r++) { m[r] = -__builtin_inff(); l[r] = 0.f; }

  const int nchunks = blockIdx.x * 4 + 4;
  for (int c = 0; c < nchunks; c++) {
    {   // stage K (async straight copy) and V (transposed, manual)
      const int key = tid >> 3, dg = (tid & 7) * 16;
      const u16* gk = Kc + (size_t)(c * 32 + key) * (NKV * HDIM) + kvh * HDIM + dg;
      u16* lk = &Ks[key * 136 + dg];
      copy16B(gk, lk);
      copy16B(gk + 8, lk + 8);

      const int key2 = tid & 31, dg2 = (tid >> 5) * 16;
      const u16* gv = Vc + (size_t)(c * 32 + key2) * (NKV * HDIM) + kvh * HDIM + dg2;
      u16 tmp[16];
      *(uint4*)&tmp[0] = ((const uint4*)gv)[0];
      *(uint4*)&tmp[8] = ((const uint4*)gv)[1];
#pragma unroll
      for (int i = 0; i < 16; i++) Vt[(dg2 + i) * LDST + key2] = tmp[i];
      async_wait_all();
    }
    __syncthreads();

    if (c * 32 <= qbase + 15) {
      v8f s0 = vz, s1 = vz;
#pragma unroll
      for (int ks = 0; ks < 4; ks++) {
        v16bf b0 = ldfrag(&Ks[lh * 136 + ks * 32 + kb]);
        v16bf b1 = ldfrag(&Ks[(16 + lh) * 136 + ks * 32 + kb]);
        s0 = wmma_bf16(qa[ks], b0, s0);
        s1 = wmma_bf16(qa[ks], b1, s1);
      }
      float alpha[8];
#pragma unroll
      for (int r = 0; r < 8; r++) {
        const int rowg = qbase + r + 8 * half;
        const int c0 = c * 32 + lh, c1 = c0 + 16;
        float v0 = s0[r] * scale; if (c0 > rowg) v0 = -__builtin_inff();
        float v1 = s1[r] * scale; if (c1 > rowg) v1 = -__builtin_inff();
        float mx = fmaxf(v0, v1);
        mx = fmaxf(mx, __shfl_xor(mx, 1, 32));
        mx = fmaxf(mx, __shfl_xor(mx, 2, 32));
        mx = fmaxf(mx, __shfl_xor(mx, 4, 32));
        mx = fmaxf(mx, __shfl_xor(mx, 8, 32));
        const float nm = fmaxf(m[r], mx);
        const float al = expf(m[r] - nm);
        m[r] = nm;
        const float p0 = expf(v0 - nm);
        const float p1 = expf(v1 - nm);
        float rs = p0 + p1;
        rs += __shfl_xor(rs, 1, 32);
        rs += __shfl_xor(rs, 2, 32);
        rs += __shfl_xor(rs, 4, 32);
        rs += __shfl_xor(rs, 8, 32);
        l[r] = l[r] * al + rs;
        alpha[r] = al;
        Ps[wid][(r + 8 * half) * LDST + lh]      = f2bf(p0);
        Ps[wid][(r + 8 * half) * LDST + 16 + lh] = f2bf(p1);
      }
#pragma unroll
      for (int nt = 0; nt < 8; nt++)
#pragma unroll
        for (int r = 0; r < 8; r++) o[nt][r] *= alpha[r];

      asm volatile("s_wait_dscnt 0" ::: "memory");  // same-wave cross-lane LDS RAW

      v16bf pa = ldfrag(&Ps[wid][lh * LDST + kb]);
#pragma unroll
      for (int nt = 0; nt < 8; nt++) {
        v16bf vb = ldfrag(&Vt[(nt * 16 + lh) * LDST + kb]);
        o[nt] = wmma_bf16(pa, vb, o[nt]);
      }
    }
    __syncthreads();
  }

#pragma unroll
  for (int nt = 0; nt < 8; nt++) {
    const int d = nt * 16 + lh;
#pragma unroll
    for (int r = 0; r < 8; r++) {
      const int rowg = qbase + r + 8 * half;
      O[(size_t)rowg * DIM + h * HDIM + d] = f2bf(o[nt][r] / l[r]);
    }
  }
}

// ---------------------------------------------------------------------------
// Supporting kernels
// ---------------------------------------------------------------------------
__global__ void k_embed(const int* __restrict__ tok, const float* __restrict__ tab,
                        float* __restrict__ x) {
  const int row = blockIdx.x;
  const float* src = tab + (size_t)tok[row] * DIM;
  float* dst = x + (size_t)row * DIM;
  for (int i = threadIdx.x; i < DIM; i += blockDim.x) dst[i] = src[i];
}

__global__ void k_rmsnorm(const float* __restrict__ x, const float* __restrict__ w,
                          u16* __restrict__ out, int dim) {
  __shared__ float red[256];
  const int row = blockIdx.x;
  const float* xr = x + (size_t)row * dim;
  float s = 0.f;
  for (int i = threadIdx.x; i < dim; i += 256) { float v = xr[i]; s += v * v; }
  red[threadIdx.x] = s;
  __syncthreads();
  for (int st = 128; st > 0; st >>= 1) {
    if (threadIdx.x < st) red[threadIdx.x] += red[threadIdx.x + st];
    __syncthreads();
  }
  const float rs = rsqrtf(red[0] / (float)dim + RMS_EPS);
  u16* orow = out + (size_t)row * dim;
  for (int i = threadIdx.x; i < dim; i += 256) orow[i] = f2bf(xr[i] * rs * w[i]);
}

__global__ void k_rope(const float* __restrict__ src, u16* __restrict__ dst,
                       int heads) {
  const int total = SEQ * heads * 64;
  for (int idx = blockIdx.x * blockDim.x + threadIdx.x; idx < total;
       idx += gridDim.x * blockDim.x) {
    const int i  = idx & 63;
    const int t  = idx >> 6;
    const int hh = t % heads;
    const int s  = t / heads;
    const float freq = powf(ROPE_THETA, -(float)i * (1.0f / 64.0f));
    const float ang = (float)s * freq;
    const float cc = cosf(ang), sn = sinf(ang);
    const size_t base = (size_t)s * heads * HDIM + (size_t)hh * HDIM + 2 * i;
    const float x0 = src[base], x1 = src[base + 1];
    dst[base]     = f2bf(x0 * cc - x1 * sn);
    dst[base + 1] = f2bf(x0 * sn + x1 * cc);
  }
}

__global__ void k_f2bf(const float* __restrict__ a, u16* __restrict__ b, int n) {
  for (int i = blockIdx.x * blockDim.x + threadIdx.x; i < n;
       i += gridDim.x * blockDim.x)
    b[i] = f2bf(a[i]);
}

__global__ void k_add(const float* __restrict__ a, const float* __restrict__ b,
                      float* __restrict__ c, int n) {
  for (int i = blockIdx.x * blockDim.x + threadIdx.x; i < n;
       i += gridDim.x * blockDim.x)
    c[i] = a[i] + b[i];
}

__global__ void k_silumul(const float* __restrict__ t1, const float* __restrict__ t3,
                          u16* __restrict__ g, int n) {
  for (int i = blockIdx.x * blockDim.x + threadIdx.x; i < n;
       i += gridDim.x * blockDim.x) {
    const float a = t1[i];
    g[i] = f2bf(a / (1.f + expf(-a)) * t3[i]);
  }
}

__global__ void k_argmax(const float* __restrict__ logits, float* __restrict__ out) {
  __shared__ float bv[256];
  __shared__ int bi[256];
  const int row = blockIdx.x;
  const float* lr = logits + (size_t)row * VOCAB;
  float best = -3.4e38f;
  int bidx = 0;
  for (int i = threadIdx.x; i < VOCAB; i += 256) {
    const float v = lr[i];
    if (v > best) { best = v; bidx = i; }
  }
  bv[threadIdx.x] = best;
  bi[threadIdx.x] = bidx;
  __syncthreads();
  for (int st = 128; st > 0; st >>= 1) {
    if (threadIdx.x < st) {
      const float ov = bv[threadIdx.x + st];
      const int oi = bi[threadIdx.x + st];
      if (ov > bv[threadIdx.x] || (ov == bv[threadIdx.x] && oi < bi[threadIdx.x])) {
        bv[threadIdx.x] = ov;
        bi[threadIdx.x] = oi;
      }
    }
    __syncthreads();
  }
  if (threadIdx.x == 0) out[row] = (float)bi[0];
}

// ---------------------------------------------------------------------------
extern "C" void kernel_launch(void* const* d_in, const int* in_sizes, int n_in,
                              void* d_out, int out_size, void* d_ws, size_t ws_size,
                              hipStream_t stream) {
  (void)in_sizes; (void)n_in; (void)out_size; (void)ws_size;
  const int*   tokens = (const int*)d_in[0];
  const float* emb    = (const float*)d_in[1];
  const float* ln1    = (const float*)d_in[2];
  const float* wq     = (const float*)d_in[3];
  const float* wk     = (const float*)d_in[4];
  const float* wv     = (const float*)d_in[5];
  const float* wo     = (const float*)d_in[6];
  const float* ln2    = (const float*)d_in[7];
  const float* w1     = (const float*)d_in[8];
  const float* w2     = (const float*)d_in[9];
  const float* w3     = (const float*)d_in[10];
  const float* lnf    = (const float*)d_in[11];
  const float* outw   = (const float*)d_in[12];

  char* ws = (char*)d_ws;
  size_t off = 0;
  auto alloc = [&](size_t bytes) {
    char* p = ws + off;
    off += (bytes + 255) & ~(size_t)255;
    return p;
  };
  float* x    = (float*)alloc((size_t)SEQ * DIM * 4);
  u16*   xn   = (u16*)  alloc((size_t)SEQ * DIM * 2);
  float* qf   = (float*)alloc((size_t)SEQ * DIM * 4);
  float* kf   = (float*)alloc((size_t)SEQ * NKV * HDIM * 4);
  float* vf   = (float*)alloc((size_t)SEQ * NKV * HDIM * 4);
  u16*   qb   = (u16*)  alloc((size_t)SEQ * DIM * 2);
  u16*   kbuf = (u16*)  alloc((size_t)SEQ * NKV * HDIM * 2);
  u16*   vbuf = (u16*)  alloc((size_t)SEQ * NKV * HDIM * 2);
  u16*   ab   = (u16*)  alloc((size_t)SEQ * DIM * 2);
  float* woo  = (float*)alloc((size_t)SEQ * DIM * 4);
  float* x2   = (float*)alloc((size_t)SEQ * DIM * 4);
  u16*   hb   = (u16*)  alloc((size_t)SEQ * DIM * 2);
  float* t1   = (float*)alloc((size_t)SEQ * FFN * 4);
  float* t3   = (float*)alloc((size_t)SEQ * FFN * 4);
  u16*   gb   = (u16*)  alloc((size_t)SEQ * FFN * 2);
  float* w2o  = (float*)alloc((size_t)SEQ * DIM * 4);
  float* fin  = (float*)alloc((size_t)SEQ * DIM * 4);
  u16*   fb   = (u16*)  alloc((size_t)SEQ * DIM * 2);

  float* nexttok = (float*)d_out;
  float* logits  = (float*)d_out + SEQ;

  k_embed<<<SEQ, 256, 0, stream>>>(tokens, emb, x);
  k_rmsnorm<<<SEQ, 256, 0, stream>>>(x, ln1, xn, DIM);

  k_gemm<true><<<dim3(DIM / 128, SEQ / 128), 256, 0, stream>>>(xn, wq, qf, SEQ, DIM, DIM);
  k_gemm<true><<<dim3((NKV * HDIM) / 128, SEQ / 128), 256, 0, stream>>>(xn, wk, kf, SEQ, NKV * HDIM, DIM);
  k_gemm<true><<<dim3((NKV * HDIM) / 128, SEQ / 128), 256, 0, stream>>>(xn, wv, vf, SEQ, NKV * HDIM, DIM);

  k_rope<<<(SEQ * NHEADS * 64 + 255) / 256, 256, 0, stream>>>(qf, qb, NHEADS);
  k_rope<<<(SEQ * NKV * 64 + 255) / 256, 256, 0, stream>>>(kf, kbuf, NKV);
  k_f2bf<<<(SEQ * NKV * HDIM + 255) / 256, 256, 0, stream>>>(vf, vbuf, SEQ * NKV * HDIM);

  k_flash<<<dim3(SEQ / 128, NHEADS), 256, 0, stream>>>(qb, kbuf, vbuf, ab);

  k_gemm<true><<<dim3(DIM / 128, SEQ / 128), 256, 0, stream>>>(ab, wo, woo, SEQ, DIM, DIM);
  k_add<<<(SEQ * DIM + 255) / 256, 256, 0, stream>>>(x, woo, x2, SEQ * DIM);

  k_rmsnorm<<<SEQ, 256, 0, stream>>>(x2, ln2, hb, DIM);
  k_gemm<false><<<dim3(FFN / 128, SEQ / 128), 256, 0, stream>>>(hb, w1, t1, SEQ, FFN, DIM);
  k_gemm<false><<<dim3(FFN / 128, SEQ / 128), 256, 0, stream>>>(hb, w3, t3, SEQ, FFN, DIM);
  k_silumul<<<(SEQ * FFN + 255) / 256, 256, 0, stream>>>(t1, t3, gb, SEQ * FFN);
  k_gemm<false><<<dim3(DIM / 128, SEQ / 128), 256, 0, stream>>>(gb, w2, w2o, SEQ, DIM, FFN);
  k_add<<<(SEQ * DIM + 255) / 256, 256, 0, stream>>>(x2, w2o, fin, SEQ * DIM);

  k_rmsnorm<<<SEQ, 256, 0, stream>>>(fin, lnf, fb, DIM);
  k_gemm<false><<<dim3(VOCAB / 128, SEQ / 128), 256, 0, stream>>>(fb, outw, logits, SEQ, VOCAB, DIM);
  k_argmax<<<SEQ, 256, 0, stream>>>(logits, nexttok);
}
